// CrossNonLocalBlockND_14224931685056
// MI455X (gfx1250) — matrix operand order repaired and checked by reading, not compiled, not run on hardware
//
#include <hip/hip_runtime.h>
#include <hip/hip_bf16.h>
#include <stdint.h>

// Problem constants (B=1)
#define NQ 18432   // D*H*W = 8*48*48
#define NK 4608    // D*(H/2)*(W/2)
#define CXC 64
#define CIC 32
#define DD 8
#define HH 48
#define WW 48
#define PHH 24
#define PWW 24
#define EPSV 1e-5f

typedef __attribute__((ext_vector_type(16))) __bf16 v16bf;
typedef __attribute__((ext_vector_type(8)))  float  v8f;
typedef __attribute__((ext_vector_type(4)))  uint32_t u32x4;
typedef __attribute__((ext_vector_type(8)))  int      i32x8;
typedef __attribute__((ext_vector_type(4)))  int      i32x4;

union Op16 { v16bf v; __bf16 e[16]; uint32_t u[8]; };
union F8   { v8f v; float f[8]; };

// Only meaningful on the device pass: __has_builtin does not report aux-target
// builtins during the host pass (the builtin calls themselves still parse there).
#if defined(__HIP_DEVICE_COMPILE__) && !__has_builtin(__builtin_amdgcn_tensor_load_to_lds)
#error "expected __builtin_amdgcn_tensor_load_to_lds on gfx1250 device toolchain"
#endif

__device__ __forceinline__ void load2x16B(Op16& op, const uint32_t* p0, const uint32_t* p1) {
    uint4 a = *(const uint4*)p0;
    uint4 b = *(const uint4*)p1;
    op.u[0] = a.x; op.u[1] = a.y; op.u[2] = a.z; op.u[3] = a.w;
    op.u[4] = b.x; op.u[5] = b.y; op.u[6] = b.z; op.u[7] = b.w;
}

// Issue one TDM load described by D# group0 (built here) + group1 (caller).
// Group0: [1:0]=count=1, [63:32]=lds_addr, [120:64]=global_addr, [127:126]=type(2=image).
__device__ __forceinline__ void tdm_issue(uint32_t lds_off, const void* gptr, i32x8 g1) {
    uint64_t ga = (uint64_t)(uintptr_t)gptr;
    u32x4 g0;
    g0[0] = 1u;                                                  // count=1, user D#
    g0[1] = lds_off;                                             // lds_addr (bytes)
    g0[2] = (uint32_t)ga;                                        // global_addr[31:0]
    g0[3] = ((uint32_t)(ga >> 32) & 0x01FFFFFFu) | (2u << 30);   // global_addr[56:32] | type=2
    i32x4 gz = {0, 0, 0, 0};
#if defined(__clang_major__) && (__clang_major__ >= 23)
    i32x8 gz8 = {0, 0, 0, 0, 0, 0, 0, 0};
    __builtin_amdgcn_tensor_load_to_lds(g0, g1, gz, gz, gz8, 0);
#else
    __builtin_amdgcn_tensor_load_to_lds(g0, g1, gz, gz, 0);
#endif
}

// ---------------- Kernel 1: Q = theta(x) -> bf16 [NQ, 32] row-major ----------------
__global__ void prep_q_kernel(const float* __restrict__ x,
                              const float* __restrict__ tw,
                              const float* __restrict__ tb,
                              __bf16* __restrict__ Qb) {
    int t = blockIdx.x * blockDim.x + threadIdx.x;
    int ci = t & 31, q = t >> 5;
    float acc = tb[ci];
    #pragma unroll 8
    for (int c = 0; c < CXC; ++c)
        acc = fmaf(tw[ci * CXC + c], x[(size_t)c * NQ + q], acc);
    Qb[(size_t)q * CIC + ci] = (__bf16)acc;
}

// ------- Kernel 2: K = pool(phi(ctx)) -> bf16 [NK,32]; V^T = pool(g(ctx)) -> bf16 [32,NK] -------
__global__ void prep_kv_kernel(const float* __restrict__ ctx,
                               const float* __restrict__ phw, const float* __restrict__ phb,
                               const float* __restrict__ gw,  const float* __restrict__ gb,
                               __bf16* __restrict__ Kb, __bf16* __restrict__ Vt) {
    int t = blockIdx.x * blockDim.x + threadIdx.x;
    int ci = t & 31, k = t >> 5;
    int d = k / (PHH * PWW);
    int rem = k - d * (PHH * PWW);
    int ph = rem / PWW;
    int pw = rem - ph * PWW;
    int i00 = d * (HH * WW) + (2 * ph) * WW + 2 * pw;
    float p0 = phb[ci], p1 = p0, p2 = p0, p3 = p0;
    float g0 = gb[ci],  g1 = g0, g2 = g0, g3 = g0;
    for (int c = 0; c < CXC; ++c) {
        float wp = phw[ci * CXC + c], wg = gw[ci * CXC + c];
        const float* base = ctx + (size_t)c * NQ + i00;
        float v0 = base[0], v1 = base[1], v2 = base[WW], v3 = base[WW + 1];
        p0 = fmaf(wp, v0, p0); p1 = fmaf(wp, v1, p1);
        p2 = fmaf(wp, v2, p2); p3 = fmaf(wp, v3, p3);
        g0 = fmaf(wg, v0, g0); g1 = fmaf(wg, v1, g1);
        g2 = fmaf(wg, v2, g2); g3 = fmaf(wg, v3, g3);
    }
    float pmax = fmaxf(fmaxf(p0, p1), fmaxf(p2, p3));
    float gmax = fmaxf(fmaxf(g0, g1), fmaxf(g2, g3));
    Kb[(size_t)k * CIC + ci] = (__bf16)pmax;
    Vt[(size_t)ci * NK + k] = (__bf16)gmax;
}

// ---------------- Kernel 3: fused flash attention (bf16 WMMA, fp32 accum) ----------------
// 8 waves/block, one 16-query tile per wave. All waves share the SAME K/V stream, so
// each 32-key chunk of K (32x64B) and V^T (32 ci rows x 64B) is staged into LDS once
// per block via the Tensor Data Mover: wave 0 issues double-buffered TDM loads that
// overlap the other waves' WMMA compute; completion is tracked with TENSORcnt.
__global__ void attn_kernel(const __bf16* __restrict__ Qb,
                            const __bf16* __restrict__ Kb,
                            const __bf16* __restrict__ Vt,
                            float* __restrict__ Y) {
    const int lane = threadIdx.x & 31;
    const int wave = threadIdx.x >> 5;
    const int qtile = blockIdx.x * (blockDim.x >> 5) + wave;
    const int qbase = qtile * 16;
    const int r = lane & 15;
    const int half = lane >> 4;
    const int hbase = half * 8;
    const int NITER = NK / 32;
    // scalar (SGPR) wave id -> TDM issue is an s_cbranch region, no EXEC juggling
    const int swave = __builtin_amdgcn_readfirstlane(threadIdx.x) >> 5;

    __shared__ __align__(16) uint32_t sK[2][512];   // 2 x 2KB: 32 K-rows x 64B
    __shared__ __align__(16) uint32_t sV[2][512];   // 2 x 2KB: 32 V^T-rows x 64B

    const uint32_t* Qu = (const uint32_t*)Qb;   // 16 dwords per Q row

    // B operand (Q^T): lane = column N = query r; dwords half*4..+3 and 8+half*4..+3
    Op16 bq;
    {
        const uint32_t* row = Qu + (size_t)(qbase + r) * 16 + half * 4;
        load2x16B(bq, row, row + 8);
    }

    F8 o0, o1;
    #pragma unroll
    for (int i = 0; i < 8; ++i) { o0.f[i] = 0.f; o1.f[i] = 0.f; }
    float m = -1e30f;
    float l = 0.f;
    v8f zero = {};

    // Static D# group1 descriptors (data_size=3 -> 8-byte units).
    // K chunk: 1-D, 256 units (2048B) contiguous.
    const i32x8 g1K = {3 << 16, 256 << 16, 0, 256 << 16, 0, 256, 0, 0};
    // V chunk: 2-D, tile 8 units x 32 rows, row stride = NK*2/8 = 1152 units.
    const i32x8 g1V = {3 << 16, 1152 << 16, 32 << 16, 8 << 16, 32, 1152, 0, 0};
    const uint32_t ldsK0 = (uint32_t)(uintptr_t)&sK[0][0];
    const uint32_t ldsK1 = (uint32_t)(uintptr_t)&sK[1][0];
    const uint32_t ldsV0 = (uint32_t)(uintptr_t)&sV[0][0];
    const uint32_t ldsV1 = (uint32_t)(uintptr_t)&sV[1][0];

    if (swave == 0) {        // prologue: DMA chunk 0 into buffer 0
        tdm_issue(ldsK0, Kb, g1K);
        tdm_issue(ldsV0, Vt, g1V);
    }

    for (int it = 0; it < NITER; ++it) {
        // Prefetch the next chunk into the other buffer. On the last iteration this
        // wraps to chunk 0 (a harmless dummy: its target buffer's readers finished
        // two barriers ago, and S_ENDPGM implicitly waits idle) so every iteration
        // uses one unconditional s_wait_tensorcnt(2) == "previous chunk complete".
        if (swave == 0) {
            int nb = (it + 1 == NITER) ? 0 : (it + 1);
            uint32_t dk = (nb & 1) ? ldsK1 : ldsK0;
            uint32_t dv = (nb & 1) ? ldsV1 : ldsV0;
            tdm_issue(dk, (const char*)Kb + (size_t)nb * 2048, g1K);
            tdm_issue(dv, (const char*)Vt + (size_t)nb * 64,   g1V);
        }
        __builtin_amdgcn_s_wait_tensorcnt(2);   // no-op for waves with TENSORcnt==0
        __syncthreads();        // staged data visible; prev compute done before reuse
        const uint32_t* kbuf = sK[it & 1];
        const uint32_t* vbuf = sV[it & 1];

        // A operands from LDS: K rows (A pattern: dwords half*4..+3 and 8+half*4..+3)
        Op16 a0, a1, bv0, bv1;
        const uint32_t* kr0 = kbuf + r * 16 + half * 4;
        load2x16B(a0, kr0, kr0 + 8);
        const uint32_t* kr1 = kbuf + (16 + r) * 16 + half * 4;
        load2x16B(a1, kr1, kr1 + 8);
        // B operands for O wmma from LDS: V^T rows (ci = column), dwords half*4..+3 and 8+half*4..+3
        const uint32_t* vr0 = vbuf + r * 16 + half * 4;
        load2x16B(bv0, vr0, vr0 + 8);
        const uint32_t* vr1 = vbuf + (16 + r) * 16 + half * 4;
        load2x16B(bv1, vr1, vr1 + 8);

        F8 st0, st1;
        st0.v = __builtin_amdgcn_wmma_f32_16x16x32_bf16(false, a0.v, false, bq.v,
                                                        (short)0, zero, false, false);
        st1.v = __builtin_amdgcn_wmma_f32_16x16x32_bf16(false, a1.v, false, bq.v,
                                                        (short)0, zero, false, false);

        // per-query running max over these 32 keys
        float mx = st0.f[0];
        #pragma unroll
        for (int i = 1; i < 8; ++i) mx = fmaxf(mx, st0.f[i]);
        #pragma unroll
        for (int i = 0; i < 8; ++i) mx = fmaxf(mx, st1.f[i]);
        mx = fmaxf(mx, __shfl_xor(mx, 16, 32));
        float mnew = fmaxf(m, mx);
        float corr = __expf(m - mnew);

        float p0[8], p1[8], s = 0.f;
        #pragma unroll
        for (int i = 0; i < 8; ++i) { p0[i] = __expf(st0.f[i] - mnew); s += p0[i]; }
        #pragma unroll
        for (int i = 0; i < 8; ++i) { p1[i] = __expf(st1.f[i] - mnew); s += p1[i]; }
        s += __shfl_xor(s, 16, 32);
        l = l * corr + s;
        m = mnew;

        // rescale O accumulators: element i lives at query row (i + 8*half)
        #pragma unroll
        for (int i = 0; i < 8; ++i) {
            float cr = __shfl(corr, hbase + i, 32);
            o0.f[i] *= cr;
            o1.f[i] *= cr;
        }

        // Pack P into A operand: C-layout of S^T already matches A layout per half
        Op16 ap;
        #pragma unroll
        for (int i = 0; i < 8; ++i) {
            ap.e[i]     = (__bf16)p0[i];
            ap.e[8 + i] = (__bf16)p1[i];
        }
        o0.v = __builtin_amdgcn_wmma_f32_16x16x32_bf16(false, ap.v, false, bv0.v,
                                                       (short)0, o0.v, false, false);
        o1.v = __builtin_amdgcn_wmma_f32_16x16x32_bf16(false, ap.v, false, bv1.v,
                                                       (short)0, o1.v, false, false);

        __syncthreads();        // all waves done with this buffer before it is refilled
    }

    float linv = 1.0f / l;
    #pragma unroll
    for (int i = 0; i < 8; ++i) {
        float li = __shfl(linv, hbase + i, 32);
        int q = qbase + i + 8 * half;
        Y[(size_t)q * CIC + r]      = o0.f[i] * li;
        Y[(size_t)q * CIC + 16 + r] = o1.f[i] * li;
    }
}

// ---------------- Kernel 4: W conv1x1 + BN partial stats ----------------
__global__ void wconv_kernel(const float* __restrict__ Y,
                             const float* __restrict__ Ww, const float* __restrict__ Wb,
                             float* __restrict__ Wy, float* __restrict__ partial) {
    int cx = blockIdx.y;
    int q = blockIdx.x * blockDim.x + threadIdx.x;
    float acc = Wb[cx];
    #pragma unroll
    for (int ci = 0; ci < CIC; ++ci)
        acc = fmaf(Ww[cx * CIC + ci], Y[(size_t)q * CIC + ci], acc);
    Wy[(size_t)cx * NQ + q] = acc;

    __shared__ float s1[256], s2[256];
    s1[threadIdx.x] = acc;
    s2[threadIdx.x] = acc * acc;
    __syncthreads();
    for (int off = 128; off > 0; off >>= 1) {
        if ((int)threadIdx.x < off) {
            s1[threadIdx.x] += s1[threadIdx.x + off];
            s2[threadIdx.x] += s2[threadIdx.x + off];
        }
        __syncthreads();
    }
    if (threadIdx.x == 0) {
        partial[(cx * gridDim.x + blockIdx.x) * 2 + 0] = s1[0];
        partial[(cx * gridDim.x + blockIdx.x) * 2 + 1] = s2[0];
    }
}

// ---------------- Kernel 5: finalize BN stats ----------------
__global__ void bnstat_kernel(const float* __restrict__ partial, int nblk,
                              float* __restrict__ stats) {
    int cx = threadIdx.x;
    float s = 0.f, ss = 0.f;
    for (int i = 0; i < nblk; ++i) {
        s  += partial[(cx * nblk + i) * 2 + 0];
        ss += partial[(cx * nblk + i) * 2 + 1];
    }
    float mean = s / (float)NQ;
    float var = ss / (float)NQ - mean * mean;
    stats[cx * 2 + 0] = mean;
    stats[cx * 2 + 1] = rsqrtf(var + EPSV);
}

// ---------------- Kernel 6: BN apply + residual ----------------
__global__ void bnapply_kernel(const float* __restrict__ Wy, const float* __restrict__ stats,
                               const float* __restrict__ gamma, const float* __restrict__ beta,
                               const float* __restrict__ x, float* __restrict__ out) {
    size_t t = (size_t)blockIdx.x * blockDim.x + threadIdx.x;
    int cx = (int)(t / NQ);
    float mu = stats[cx * 2 + 0], rstd = stats[cx * 2 + 1];
    out[t] = (Wy[t] - mu) * rstd * gamma[cx] + beta[cx] + x[t];
}

extern "C" void kernel_launch(void* const* d_in, const int* in_sizes, int n_in,
                              void* d_out, int out_size, void* d_ws, size_t ws_size,
                              hipStream_t stream) {
    (void)in_sizes; (void)n_in; (void)out_size; (void)ws_size;
    const float* x       = (const float*)d_in[0];
    const float* context = (const float*)d_in[1];
    const float* theta_w = (const float*)d_in[2];
    const float* theta_b = (const float*)d_in[3];
    const float* phi_w   = (const float*)d_in[4];
    const float* phi_b   = (const float*)d_in[5];
    const float* g_w     = (const float*)d_in[6];
    const float* g_b     = (const float*)d_in[7];
    const float* W_w     = (const float*)d_in[8];
    const float* W_b     = (const float*)d_in[9];
    const float* bn_g    = (const float*)d_in[10];
    const float* bn_b    = (const float*)d_in[11];
    float* out = (float*)d_out;

    char* ws = (char*)d_ws;
    __bf16* Qb     = (__bf16*)(ws + 0);              // NQ*32*2  = 1179648 B
    __bf16* Kb     = (__bf16*)(ws + 1179648);        // NK*32*2  =  294912 B
    __bf16* Vt     = (__bf16*)(ws + 1474560);        // 32*NK*2  =  294912 B
    float*  Y      = (float*) (ws + 1769472);        // NQ*32*4  = 2359296 B
    float*  Wy     = (float*) (ws + 4128768);        // 64*NQ*4  = 4718592 B
    float*  part   = (float*) (ws + 8847360);        // 64*72*2*4 = 36864 B
    float*  stats  = (float*) (ws + 8884224);        // 64*2*4

    // 1) Q projection
    prep_q_kernel<<<(NQ * CIC) / 256, 256, 0, stream>>>(x, theta_w, theta_b, Qb);
    // 2) K, V projections + maxpool
    prep_kv_kernel<<<(NK * CIC) / 256, 256, 0, stream>>>(context, phi_w, phi_b, g_w, g_b, Kb, Vt);
    // 3) fused attention: 1152 query tiles, 8 waves/block
    attn_kernel<<<(NQ / 16) / 8, 256, 0, stream>>>(Qb, Kb, Vt, Y);
    // 4) W conv + BN partials
    dim3 wgrid(NQ / 256, CXC);
    wconv_kernel<<<wgrid, 256, 0, stream>>>(Y, W_w, W_b, Wy, part);
    // 5) BN stats
    bnstat_kernel<<<1, CXC, 0, stream>>>(part, NQ / 256, stats);
    // 6) BN apply + residual
    bnapply_kernel<<<(CXC * NQ) / 256, 256, 0, stream>>>(Wy, stats, bn_g, bn_b, x, out);
}
